// KernelEncoder_19207093748412
// MI455X (gfx1250) — matrix operand order repaired and checked by reading, not compiled, not run on hardware
//
#include <hip/hip_runtime.h>
#include <hip/hip_bf16.h>

// ---------------------------------------------------------------------------
// Performer (kernel-attention) encoder layer for MI455X / gfx1250.
// All contractions run through v_wmma_f32_16x16x32_f16 (f16 in, f32 acc).
// Inner loop: swizzled LDS B-staging (fragment-order, ds_load_b128 reads) and
// 2 column tiles per wave so each A fragment feeds 2 WMMAs.
// ---------------------------------------------------------------------------

typedef __attribute__((ext_vector_type(16))) _Float16 v16h;
typedef __attribute__((ext_vector_type(8)))  _Float16 v8h;
typedef __attribute__((ext_vector_type(4)))  _Float16 v4h;
typedef __attribute__((ext_vector_type(8)))  float    v8f;

#define EPS_LN 1e-3f
#define STAB   1e-6f

constexpr int Bb = 4, Ss = 4096, Dd = 128, Hh = 8, Kk = 128, Mm = 256;
constexpr int BS = Bb * Ss;     // 16384 rows
constexpr int HK = Hh * Kk;     // 1024
constexpr int HM = Hh * Mm;     // 2048

// epilogue modes
constexpr int EP_NONE = 0;   // (acc + bias) * scale
constexpr int EP_ELU1 = 1;   // (elu(acc + bias) + 1) * scale   (Performer feature map)
constexpr int EP_ELU  = 2;   // elu(acc + bias) * scale         (FFN activation)
// output scatter layouts
constexpr int OL_ROW    = 0; // z*cBatch + row*ldc + col
constexpr int OL_BHSK   = 1; // [B,H,S,K] from (row=b*S+s, col=h*K+c)
constexpr int OL_Z_BHSM = 2; // [B,H,S,M] from (row=b*S+s, col=m, z=h)
constexpr int OL_Z_T    = 3; // [B,H,M,S] transposed store (row=b*S+s, col=m, z=h)
constexpr int OL_Z_BSHK = 4; // [B,S,H,K] from (row=s, col=k, z=b*H+h)

// ---------------------------------------------------------------------------
// Generic WMMA GEMM: C = epilogue(A[f16] x B + bias).
// Block = 256 threads = 8 waves; each wave computes a 16x32 output strip
// (2 WMMA tiles sharing one A fragment); block covers 128 rows x 32 cols.
// grid = (Mrows/128, N/32, batches).
// B slab is staged into LDS in WMMA B-fragment order: row (nt*32 + lane)
// holds that lane's 16 K-halves (k = 8h..8h+7, 16+8h..16+8h+7), rows padded to 24 halves (48 B)
// so every lane chunk is 16B-aligned -> fragment read = 2x ds_load_b128.
// ---------------------------------------------------------------------------
template <bool B_F16, bool B_TRANS, int EP, int OL,
          bool HAS_BIAS, bool HAS_RS, bool HAS_RESID, bool OUT_F16>
__global__ __launch_bounds__(256) void wmma_gemm_kernel(
    const _Float16* __restrict__ A, long long aBatch, int lda,
    const void* __restrict__ Bp, long long bBatch, int ldb,
    const float* __restrict__ bias,
    const float* __restrict__ rowscale, long long rsBatch,
    const float* __restrict__ resid, int ldresid,
    void* __restrict__ Cp, long long cBatch, int ldc,
    float scale, int Kdim, int S, int H, int Nfull)
{
  constexpr int NT = 2;                 // column tiles per wave
  __shared__ _Float16 Bsw[NT * 32][24]; // fragment-order B slab, padded rows

  const int tid  = threadIdx.x;
  const int wave = tid >> 5;
  const int lane = tid & 31;
  const int ml   = lane & 15;           // row (A) / col (B,C) within tile
  const int hi   = lane >> 4;           // lane-half selector
  const int z    = blockIdx.z;
  const int rowbase = blockIdx.x * 128 + wave * 16;
  const int colbase = blockIdx.y * 32;

  const _Float16* Arow =
      A + (long long)z * aBatch + (size_t)(rowbase + ml) * lda + 8 * hi;

  v8f acc[NT] = {};
  for (int k0 = 0; k0 < Kdim; k0 += 32) {
    // speculative prefetch of the next A chunk (dropped safely if OOB)
    __builtin_prefetch(Arow + k0 + 32, 0, 1);

    // --- stage B[k0..k0+31][colbase..+31] into LDS in fragment order
    {
      int d = tid;
#pragma unroll
      for (int it = 0; it < 4; ++it, d += 256) {
        const int n2 = d & 31;          // column within 32-wide block strip
        const int kk = d >> 5;          // 0..31 within K chunk
        const long long gk = k0 + kk, gn = colbase + n2;
        float v;
        if constexpr (B_F16) {
          const _Float16* Bh = (const _Float16*)Bp;
          v = (float)(B_TRANS ? Bh[(long long)z * bBatch + gn * (long long)ldb + gk]
                              : Bh[(long long)z * bBatch + gk * (long long)ldb + gn]);
        } else {
          const float* Bf = (const float*)Bp;
          v = B_TRANS ? Bf[(long long)z * bBatch + gn * (long long)ldb + gk]
                      : Bf[(long long)z * bBatch + gk * (long long)ldb + gn];
        }
        // fragment position: tile = n2>>4, lane = ((kk>>3)&1)*16 + (n2&15), j = (kk&7) + 8*(kk>>4)
        Bsw[(n2 >> 4) * 32 + ((kk >> 3) & 1) * 16 + (n2 & 15)][(kk & 7) + 8 * (kk >> 4)] = (_Float16)v;
      }
    }
    __syncthreads();

    // --- A fragment: 16-bit A 16x32 layout (ISA 7.12.2), two 16B chunks/lane
    const v8h alo = *(const v8h*)(Arow + k0);        // K = 8*hi .. +7
    const v8h ahi = *(const v8h*)(Arow + k0 + 16);   // K = 16+8*hi .. +7
    v16h a;
#pragma unroll
    for (int j = 0; j < 8; ++j) { a[j] = alo[j]; a[8 + j] = ahi[j]; }

    // --- 2 WMMAs sharing the A fragment; B fragment = 2x ds_load_b128
#pragma unroll
    for (int nt = 0; nt < NT; ++nt) {
      const v8h blo = *(const v8h*)&Bsw[nt * 32 + lane][0];
      const v8h bhi = *(const v8h*)&Bsw[nt * 32 + lane][8];
      v16h b;
#pragma unroll
      for (int j = 0; j < 8; ++j) { b[j] = blo[j]; b[8 + j] = bhi[j]; }
      acc[nt] = __builtin_amdgcn_wmma_f32_16x16x32_f16(
          false, a, false, b, (short)0, acc[nt], false, false);
    }
    __syncthreads();
  }

  // --- epilogue + scatter store (C layout: lane col = ml, VGPR r -> M=r+8*hi)
  _Float16* C16 = (_Float16*)Cp;
  float*    C32 = (float*)Cp;
#pragma unroll
  for (int nt = 0; nt < NT; ++nt) {
    const int col = colbase + nt * 16 + ml;
    const float bvv = HAS_BIAS ? bias[col] : 0.0f;
#pragma unroll
    for (int r = 0; r < 8; ++r) {
      const int row = rowbase + r + 8 * hi;
      float x = acc[nt][r] + bvv;
      if constexpr (EP == EP_ELU1)     x = (x > 0.0f ? x + 1.0f : __expf(x)) * scale;
      else if constexpr (EP == EP_ELU) x = (x > 0.0f ? x : __expf(x) - 1.0f) * scale;
      else                             x = x * scale;
      if constexpr (HAS_RS)    x *= rowscale[(long long)z * rsBatch + row];
      if constexpr (HAS_RESID) x += resid[(size_t)row * ldresid + col];

      size_t idx;
      if constexpr (OL == OL_ROW) {
        idx = (size_t)((long long)z * cBatch) + (size_t)row * ldc + col;
      } else if constexpr (OL == OL_BHSK) {
        const int bb = row / S, sl = row - bb * S;
        const int hh = col / ldc, cc = col - hh * ldc;
        idx = (((size_t)(bb * H + hh)) * S + sl) * (size_t)ldc + cc;
      } else if constexpr (OL == OL_Z_BHSM) {
        const int bb = row / S, sl = row - bb * S;
        idx = (((size_t)(bb * H + z)) * S + sl) * (size_t)Nfull + col;
      } else if constexpr (OL == OL_Z_T) {
        const int bb = row / S, sl = row - bb * S;
        idx = (((size_t)(bb * H + z)) * Nfull + col) * (size_t)S + sl;
      } else { // OL_Z_BSHK
        const int bb = z / H, hh = z - bb * H;
        idx = (((size_t)bb * S + row) * H + hh) * (size_t)ldc + col;
      }
      if constexpr (OUT_F16) C16[idx] = (_Float16)x;
      else                   C32[idx] = x;
    }
  }
}

// ---------------------------------------------------------------------------
// LayerNorm over D=128: one wave per row, butterfly reduction (wave32).
// ---------------------------------------------------------------------------
__global__ __launch_bounds__(256) void layernorm128_kernel(
    const float* __restrict__ in, const float* __restrict__ gam,
    const float* __restrict__ bet, float* __restrict__ out32,
    _Float16* __restrict__ out16, int nrows)
{
  const int wave = threadIdx.x >> 5;
  const int lane = threadIdx.x & 31;
  const int row  = blockIdx.x * 8 + wave;
  if (row >= nrows) return;
  const float4 x = *(const float4*)(in + (size_t)row * 128 + lane * 4);
  float s  = x.x + x.y + x.z + x.w;
  float ss = x.x * x.x + x.y * x.y + x.z * x.z + x.w * x.w;
#pragma unroll
  for (int m = 16; m >= 1; m >>= 1) {
    s  += __shfl_xor(s,  m, 32);
    ss += __shfl_xor(ss, m, 32);
  }
  const float mean = s * (1.0f / 128.0f);
  const float var  = ss * (1.0f / 128.0f) - mean * mean;
  const float rstd = rsqrtf(var + EPS_LN);
  const int c = lane * 4;
  const float4 g = *(const float4*)(gam + c);
  const float4 b = *(const float4*)(bet + c);
  const float y0 = (x.x - mean) * rstd * g.x + b.x;
  const float y1 = (x.y - mean) * rstd * g.y + b.y;
  const float y2 = (x.z - mean) * rstd * g.z + b.z;
  const float y3 = (x.w - mean) * rstd * g.w + b.w;
  if (out32) {
    float4 o; o.x = y0; o.y = y1; o.z = y2; o.w = y3;
    *(float4*)(out32 + (size_t)row * 128 + c) = o;
  }
  if (out16) {
    v4h o = {(_Float16)y0, (_Float16)y1, (_Float16)y2, (_Float16)y3};
    *(v4h*)(out16 + (size_t)row * 128 + c) = o;
  }
}

__global__ __launch_bounds__(256) void f32_to_f16x4_kernel(
    const float* __restrict__ in, _Float16* __restrict__ out, int n)
{
  const int i = blockIdx.x * 256 + threadIdx.x;
  if (i * 4 >= n) return;
  const float4 x = ((const float4*)in)[i];
  v4h o = {(_Float16)x.x, (_Float16)x.y, (_Float16)x.z, (_Float16)x.w};
  ((v4h*)out)[i] = o;
}

__global__ __launch_bounds__(256) void i32_to_f32_kernel(
    const int* __restrict__ in, float* __restrict__ out, int n)
{
  const int i = blockIdx.x * 256 + threadIdx.x;
  if (i < n) out[i] = (float)in[i];
}

// ksum[b,h,m] = sum_s kp_t[b,h,m,s] ; kp_t is [B,H,M,S] f16 (contiguous in s)
__global__ __launch_bounds__(256) void ksum_kernel(
    const _Float16* __restrict__ kpt, float* __restrict__ ksum, int S)
{
  const int i = blockIdx.x * 256 + threadIdx.x;   // over B*H*M
  const _Float16* p = kpt + (size_t)i * S;
  float acc = 0.0f;
  for (int s = 0; s < S; s += 8) {
    const v8h v = *(const v8h*)(p + s);
#pragma unroll
    for (int j = 0; j < 8; ++j) acc += (float)v[j];
  }
  ksum[i] = acc;
}

// denom[b,h,t] = 1 / (qp[b,h,t,:] . ksum[b,h,:] + STAB)
__global__ __launch_bounds__(256) void denom_kernel(
    const _Float16* __restrict__ qp, const float* __restrict__ ksum,
    float* __restrict__ denom, int S, int M)
{
  const int i  = blockIdx.x * 256 + threadIdx.x;  // over B*H*S
  const int bh = i / S;
  const _Float16* q = qp + (size_t)i * M;
  const float* ks = ksum + (size_t)bh * M;
  float acc = 0.0f;
  for (int m = 0; m < M; m += 8) {
    const v8h v = *(const v8h*)(q + m);
#pragma unroll
    for (int j = 0; j < 8; ++j) acc += (float)v[j] * ks[m + j];
  }
  denom[i] = 1.0f / (acc + STAB);
}

// ---------------------------------------------------------------------------
extern "C" void kernel_launch(void* const* d_in, const int* in_sizes, int n_in,
                              void* d_out, int out_size, void* d_ws, size_t ws_size,
                              hipStream_t stream)
{
  (void)in_sizes; (void)n_in; (void)out_size; (void)ws_size;

  const float* Q      = (const float*)d_in[0];
  const float* X      = (const float*)d_in[1];
  const int*   mask   = (const int*)  d_in[2];
  const float* Wq     = (const float*)d_in[3];
  const float* bq     = (const float*)d_in[4];
  const float* Wk     = (const float*)d_in[5];
  const float* bk     = (const float*)d_in[6];
  const float* Wv     = (const float*)d_in[7];
  const float* bv     = (const float*)d_in[8];
  const float* Wo     = (const float*)d_in[9];
  const float* bo     = (const float*)d_in[10];
  const float* proj   = (const float*)d_in[11];
  const float* ln1_g  = (const float*)d_in[12];
  const float* ln1_b  = (const float*)d_in[13];
  const float* ln2_g  = (const float*)d_in[14];
  const float* ln2_b  = (const float*)d_in[15];
  const float* fln0_g = (const float*)d_in[16];
  const float* fln0_b = (const float*)d_in[17];
  const float* f_w0   = (const float*)d_in[18];
  const float* f_b0   = (const float*)d_in[19];
  const float* fln1_g = (const float*)d_in[20];
  const float* fln1_b = (const float*)d_in[21];
  const float* f_w1   = (const float*)d_in[22];
  const float* f_b1   = (const float*)d_in[23];
  float* out = (float*)d_out;

  // ----- workspace layout (~254 MB, 256B aligned) -----
  char* p = (char*)d_ws;
  auto alloc = [&](size_t bytes) -> char* {
    char* r = p; p += (bytes + 255) & ~(size_t)255; return r;
  };
  float*    mskf = (float*)   alloc((size_t)BS * 4);
  float*    Xn   = (float*)   alloc((size_t)BS * Dd * 4);
  _Float16* Xn_h = (_Float16*)alloc((size_t)BS * Dd * 2);
  _Float16* Q_h  = (_Float16*)alloc((size_t)BS * Dd * 2);
  _Float16* q_h  = (_Float16*)alloc((size_t)BS * HK * 2);
  char*     kblk =            alloc((size_t)BS * HK * 2);   // 32 MB, reused later
  _Float16* k_h  = (_Float16*)kblk;
  _Float16* v_t  = (_Float16*)alloc((size_t)BS * HK * 2);   // [B,H,S,K]
  _Float16* qp   = (_Float16*)alloc((size_t)BS * HM * 2);   // [B,H,S,M]
  _Float16* kp_t = (_Float16*)alloc((size_t)BS * HM * 2);   // [B,H,M,S]
  float*    ksum = (float*)   alloc((size_t)Bb * Hh * Mm * 4);
  _Float16* kv_h = (_Float16*)alloc((size_t)Bb * Hh * Mm * Kk * 2);
  float*    denm = (float*)   alloc((size_t)Bb * Hh * Ss * 4);
  // aliases (dead buffers reused)
  _Float16* attn_h = q_h;                             // q consumed by qp GEMM
  float*    res1 = (float*)kblk;                      // k consumed by kp GEMM
  float*    out1 = (float*)(kblk + 8388608);
  float*    h1   = (float*)(kblk + 16777216);
  _Float16* ln0h = (_Float16*)(kblk + 25165824);
  _Float16* ln1h = (_Float16*)(kblk + 29360128);

  const float inv_sqrt_k = 0.08838834764831845f;  // 1/sqrt(128)
  const float inv_sqrt_m = 0.0625f;               // 1/sqrt(256)

  // 0) mask -> float ; LN1(X) -> Xn (f32 + f16) ; Q -> f16
  i32_to_f32_kernel<<<BS / 256, 256, 0, stream>>>(mask, mskf, BS);
  layernorm128_kernel<<<BS / 8, 256, 0, stream>>>(X, ln1_g, ln1_b, Xn, Xn_h, BS);
  f32_to_f16x4_kernel<<<(BS * Dd / 4) / 256, 256, 0, stream>>>(Q, Q_h, BS * Dd);

  // 1) q = (Q @ Wq + bq) * 1/sqrt(K)   -> f16 [BS,HK]
  wmma_gemm_kernel<false, false, EP_NONE, OL_ROW, true, false, false, true>
      <<<dim3(BS / 128, HK / 32, 1), 256, 0, stream>>>(
          Q_h, 0, Dd, Wq, 0, HK, bq, nullptr, 0, nullptr, 0,
          q_h, 0, HK, inv_sqrt_k, Dd, Ss, Hh, HK);
  // 2) k = Xn @ Wk + bk                 -> f16 [BS,HK]
  wmma_gemm_kernel<false, false, EP_NONE, OL_ROW, true, false, false, true>
      <<<dim3(BS / 128, HK / 32, 1), 256, 0, stream>>>(
          Xn_h, 0, Dd, Wk, 0, HK, bk, nullptr, 0, nullptr, 0,
          k_h, 0, HK, 1.0f, Dd, Ss, Hh, HK);
  // 3) v = Xn @ Wv + bv                 -> f16 [B,H,S,K] (scatter)
  wmma_gemm_kernel<false, false, EP_NONE, OL_BHSK, true, false, false, true>
      <<<dim3(BS / 128, HK / 32, 1), 256, 0, stream>>>(
          Xn_h, 0, Dd, Wv, 0, HK, bv, nullptr, 0, nullptr, 0,
          v_t, 0, Kk, 1.0f, Dd, Ss, Hh, HK);

  // 4) qp = (elu(q_h @ projT)+1)/sqrt(M)        -> f16 [B,H,S,M], batched per head
  wmma_gemm_kernel<false, true, EP_ELU1, OL_Z_BHSM, false, false, false, true>
      <<<dim3(BS / 128, Mm / 32, Hh), 256, 0, stream>>>(
          q_h, Kk, HK, proj, 0, Kk, nullptr, nullptr, 0, nullptr, 0,
          qp, 0, Mm, inv_sqrt_m, Kk, Ss, Hh, Mm);
  // 5) kp = (elu(k_h @ projT)+1)/sqrt(M)*mask   -> f16 [B,H,M,S] (transposed)
  wmma_gemm_kernel<false, true, EP_ELU1, OL_Z_T, false, true, false, true>
      <<<dim3(BS / 128, Mm / 32, Hh), 256, 0, stream>>>(
          k_h, Kk, HK, proj, 0, Kk, nullptr, mskf, 0, nullptr, 0,
          kp_t, 0, Mm, inv_sqrt_m, Kk, Ss, Hh, Mm);

  // 6) ksum[b,h,m] = sum_s kp
  ksum_kernel<<<(Bb * Hh * Mm) / 256, 256, 0, stream>>>(kp_t, ksum, Ss);
  // 7) kv = kp^T @ v : [M,S]x[S,K] per (b,h)     -> f16 [B,H,M,K]
  wmma_gemm_kernel<true, false, EP_NONE, OL_ROW, false, false, false, true>
      <<<dim3(Mm / 128, Kk / 32, Bb * Hh), 256, 0, stream>>>(
          kp_t, (long long)Mm * Ss, Ss, v_t, (long long)Ss * Kk, Kk,
          nullptr, nullptr, 0, nullptr, 0,
          kv_h, (long long)Mm * Kk, Kk, 1.0f, Ss, Ss, Hh, Kk);
  // 8) denom = 1/(qp . ksum + STAB)              -> f32 [B,H,S]
  denom_kernel<<<(Bb * Hh * Ss) / 256, 256, 0, stream>>>(qp, ksum, denm, Ss, Mm);
  // 9) attn = (qp @ kv) * denom                  -> f16 [B,S,H,K] (scatter)
  wmma_gemm_kernel<true, false, EP_NONE, OL_Z_BSHK, false, true, false, true>
      <<<dim3(Ss / 128, Kk / 32, Bb * Hh), 256, 0, stream>>>(
          qp, (long long)Ss * Mm, Mm, kv_h, (long long)Mm * Kk, Kk,
          nullptr, denm, Ss, nullptr, 0,
          attn_h, 0, Kk, 1.0f, Mm, Ss, Hh, Kk);

  // 10) res1 = (attn @ Wo + bo)*mask + Xn        -> f32 [BS,D]
  wmma_gemm_kernel<false, false, EP_NONE, OL_ROW, true, true, true, false>
      <<<dim3(BS / 128, Dd / 32, 1), 256, 0, stream>>>(
          attn_h, 0, HK, Wo, 0, Dd, bo, mskf, 0, Xn, Dd,
          res1, 0, Dd, 1.0f, HK, Ss, Hh, Dd);

  // 11) out1 = LN2(res1);  ln0h = LN(out1, f_ln0) as f16
  layernorm128_kernel<<<BS / 8, 256, 0, stream>>>(res1, ln2_g, ln2_b, out1, nullptr, BS);
  layernorm128_kernel<<<BS / 8, 256, 0, stream>>>(out1, fln0_g, fln0_b, nullptr, ln0h, BS);
  // 12) h1 = elu(ln0h @ f_w0 + f_b0)             -> f32
  wmma_gemm_kernel<false, false, EP_ELU, OL_ROW, true, false, false, false>
      <<<dim3(BS / 128, Dd / 32, 1), 256, 0, stream>>>(
          ln0h, 0, Dd, f_w0, 0, Dd, f_b0, nullptr, 0, nullptr, 0,
          h1, 0, Dd, 1.0f, Dd, Ss, Hh, Dd);
  // 13) ln1h = LN(h1, f_ln1) as f16
  layernorm128_kernel<<<BS / 8, 256, 0, stream>>>(h1, fln1_g, fln1_b, nullptr, ln1h, BS);
  // 14) out = ln1h @ f_w1 + f_b1                 -> f32 [B,S,D]
  wmma_gemm_kernel<false, false, EP_NONE, OL_ROW, true, false, false, false>
      <<<dim3(BS / 128, Dd / 32, 1), 256, 0, stream>>>(
          ln1h, 0, Dd, f_w1, 0, Dd, f_b1, nullptr, 0, nullptr, 0,
          out, 0, Dd, 1.0f, Dd, Ss, Hh, Dd);
}
